// GAT_41721312313368
// MI455X (gfx1250) — matrix operand order repaired and checked by reading, not compiled, not run on hardware
//
#include <hip/hip_runtime.h>

// ---------------------------------------------------------------------------
// GAT-style message passing on gfx1250 (MI455X).
// All matmuls on v_wmma_f32_16x16x32_bf16 (fp32 accumulate), softmax stats in
// fp32, scores/probabilities staged in LDS as bf16.
// Update fusion: relu((msg+self) @ W^T + 2b).
// ---------------------------------------------------------------------------

#define DD 256  // feature dim

typedef __attribute__((ext_vector_type(16))) __bf16 v16bf;
typedef __attribute__((ext_vector_type(8)))  float  v8f;

static __device__ inline v8f wmma_bf16(v16bf a, v16bf b, v8f c) {
  // (neg_a, A, neg_b, B, c_mod, C, reuse_a, reuse_b)
  return __builtin_amdgcn_wmma_f32_16x16x32_bf16(false, a, false, b,
                                                 (short)0, c, false, false);
}

// A fragment (16x32, row-major source, leading dim ld), ISA 16-bit A layout:
// lanes 0-15: row=lane, K = 0..7 then 16..23 ; lanes 16-31: row=lane-16,
// K = 8..15 then 24..31.
static __device__ inline v16bf frag_a_f32(const float* __restrict__ base, int ld) {
  const int lane = threadIdx.x & 31;
  const int r = lane & 15;
  const int kb = (lane < 16) ? 0 : 8;
  const float4* p = (const float4*)(base + (size_t)r * ld + kb);
  const float4 f0 = p[0];           // K kb+0..3
  const float4 f1 = p[1];           // K kb+4..7
  const float4 f2 = p[4];           // K kb+16..19
  const float4 f3 = p[5];           // K kb+20..23
  v16bf v;
  v[0]  = (__bf16)f0.x; v[1]  = (__bf16)f0.y; v[2]  = (__bf16)f0.z; v[3]  = (__bf16)f0.w;
  v[4]  = (__bf16)f1.x; v[5]  = (__bf16)f1.y; v[6]  = (__bf16)f1.z; v[7]  = (__bf16)f1.w;
  v[8]  = (__bf16)f2.x; v[9]  = (__bf16)f2.y; v[10] = (__bf16)f2.z; v[11] = (__bf16)f2.w;
  v[12] = (__bf16)f3.x; v[13] = (__bf16)f3.y; v[14] = (__bf16)f3.z; v[15] = (__bf16)f3.w;
  return v;
}

static __device__ inline v16bf frag_a_bf16(const __bf16* base, int ld) {
  const int lane = threadIdx.x & 31;
  const int r = lane & 15;
  const int kb = (lane < 16) ? 0 : 8;
  union { v16bf v; uint4 q[2]; } u;
  u.q[0] = *(const uint4*)(base + (size_t)r * ld + kb);
  u.q[1] = *(const uint4*)(base + (size_t)r * ld + kb + 16);
  return u.v;
}

// B fragment (32x16). Source is "B-transposed" row-major storage: column n of
// the B operand is 32 contiguous K-values at row n of the source tile.
// Lanes 0-15: col=lane, K=0..7 then 16..23 ; lanes 16-31: col=lane-16, K=8..15 then 24..31.
static __device__ inline v16bf frag_b_f32(const float* __restrict__ base, int ld) {
  const int lane = threadIdx.x & 31;
  const int n = lane & 15;
  const int kb = (lane < 16) ? 0 : 8;
  const float4* p = (const float4*)(base + (size_t)n * ld + kb);
  const float4 f0 = p[0];
  const float4 f1 = p[1];
  const float4 f2 = p[4];
  const float4 f3 = p[5];
  v16bf v;
  v[0]  = (__bf16)f0.x; v[1]  = (__bf16)f0.y; v[2]  = (__bf16)f0.z; v[3]  = (__bf16)f0.w;
  v[4]  = (__bf16)f1.x; v[5]  = (__bf16)f1.y; v[6]  = (__bf16)f1.z; v[7]  = (__bf16)f1.w;
  v[8]  = (__bf16)f2.x; v[9]  = (__bf16)f2.y; v[10] = (__bf16)f2.z; v[11] = (__bf16)f2.w;
  v[12] = (__bf16)f3.x; v[13] = (__bf16)f3.y; v[14] = (__bf16)f3.z; v[15] = (__bf16)f3.w;
  return v;
}

static __device__ inline v16bf frag_b_bf16(const __bf16* base, int ld) {
  const int lane = threadIdx.x & 31;
  const int n = lane & 15;
  const int kb = (lane < 16) ? 0 : 8;
  union { v16bf v; uint4 q[2]; } u;
  u.q[0] = *(const uint4*)(base + (size_t)n * ld + kb);
  u.q[1] = *(const uint4*)(base + (size_t)n * ld + kb + 16);
  return u.v;
}

// ---------------------------------------------------------------------------
// Y[rows,DD] = op(X[rows,DD] @ W^T + bscale*bias), W row-major [out,in].
// 4 waves / block; each wave: 16 rows x 64 cols, K=256 in 8 wmma steps,
// A-fragment reused across 4 B-fragments.
// ---------------------------------------------------------------------------
template <bool OUT_BF16, bool RELU>
__global__ __launch_bounds__(128) void linear_wmma(
    const float* __restrict__ X, const float* __restrict__ W,
    const float* __restrict__ bias, float bscale,
    float* __restrict__ Yf, __bf16* __restrict__ Yb, int rows)
{
  const int wave = threadIdx.x >> 5;
  const int lane = threadIdx.x & 31;
  const int row0 = blockIdx.x * 64 + wave * 16;
  const int col0 = blockIdx.y * 64;
  if (row0 >= rows) return;  // uniform per wave

  v8f acc[4] = {v8f{}, v8f{}, v8f{}, v8f{}};
  for (int k = 0; k < DD; k += 32) {
    v16bf a = frag_a_f32(X + (size_t)row0 * DD + k, DD);
#pragma unroll
    for (int c = 0; c < 4; ++c) {
      v16bf b = frag_b_f32(W + (size_t)(col0 + 16 * c) * DD + k, DD);
      acc[c] = wmma_bf16(a, b, acc[c]);
    }
  }
  const int hi = lane >> 4;
  const int cn = lane & 15;
#pragma unroll
  for (int c = 0; c < 4; ++c) {
    const int col = col0 + 16 * c + cn;
    const float bv = bscale * bias[col];
#pragma unroll
    for (int i = 0; i < 8; ++i) {
      const size_t r = (size_t)(row0 + i + 8 * hi);
      float v = acc[c][i] + bv;
      if (RELU) v = v > 0.f ? v : 0.f;
      if (OUT_BF16) Yb[r * DD + col] = (__bf16)v;
      else          Yf[r * DD + col] = v;
    }
  }
}

// ---------------------------------------------------------------------------
// Masked attention, one 16-query tile per block (4 waves cooperate).
//   S[q,key] = (Q[q,:] . K[key,:]) * mask[b,key,q]   (multiplicative mask)
//   P = softmax over keys (row-wise); out[q,:] = P @ V, V given transposed.
// Scores/probabilities live in LDS as bf16 with a +8 row pad (4-bank rotation
// per row -> conflict-free 16-row fragment loads).
// ---------------------------------------------------------------------------
__global__ __launch_bounds__(128) void attn_wmma(
    const __bf16* __restrict__ Q, const __bf16* __restrict__ K,
    const __bf16* __restrict__ Vt, const float* __restrict__ mask,
    float* __restrict__ out, int nQ, int nK)
{
  __shared__ __align__(16) __bf16 S[16 * (2048 + 8)];
  __shared__ float red[16][8];
  __shared__ float rowmax[16];
  __shared__ float rowinv[16];

  const int ldS  = nK + 8;
  const int b    = blockIdx.y;
  const int q0   = blockIdx.x * 16;
  const int wave = threadIdx.x >> 5;
  const int lane = threadIdx.x & 31;

  const __bf16* Qb = Q + ((size_t)b * nQ + q0) * DD;
  const __bf16* Kb = K + (size_t)b * nK * DD;

  // Phase 1: scores -> masked -> bf16 into LDS. Waves split key tiles.
  for (int kt = wave; kt < nK / 16; kt += 4) {
    v8f acc = v8f{};
    for (int k = 0; k < DD; k += 32) {
      v16bf a  = frag_a_bf16(Qb + k, DD);
      v16bf bb = frag_b_bf16(Kb + (size_t)kt * 16 * DD + k, DD);
      acc = wmma_bf16(a, bb, acc);
    }
    const int hi  = lane >> 4;
    const int key = kt * 16 + (lane & 15);
    const float* mcol = mask + ((size_t)b * nK + key) * nQ + q0;  // [b,key,q]
#pragma unroll
    for (int i = 0; i < 8; ++i) {
      const int r = i + 8 * hi;
      S[r * ldS + key] = (__bf16)(acc[i] * mcol[r]);
    }
  }
  __syncthreads();

  // Phase 2: row softmax stats; store unnormalized exp back in place (bf16),
  // fold 1/sum into the output epilogue.
  {
    const int r = threadIdx.x >> 3;  // 16 rows
    const int s = threadIdx.x & 7;   // 8 threads per row
    float m = -3.0e38f;
    for (int j = s; j < nK; j += 8) m = fmaxf(m, (float)S[r * ldS + j]);
    red[r][s] = m;
    __syncthreads();
    if (s == 0) {
      float mm = red[r][0];
#pragma unroll
      for (int t = 1; t < 8; ++t) mm = fmaxf(mm, red[r][t]);
      rowmax[r] = mm;
    }
    __syncthreads();
    const float mm = rowmax[r];
    float sum = 0.f;
    for (int j = s; j < nK; j += 8) {
      const float e = __expf((float)S[r * ldS + j] - mm);
      S[r * ldS + j] = (__bf16)e;
      sum += e;
    }
    red[r][s] = sum;
    __syncthreads();
    if (s == 0) {
      float ss = 0.f;
#pragma unroll
      for (int t = 0; t < 8; ++t) ss += red[r][t];
      rowinv[r] = 1.0f / ss;
    }
  }
  __syncthreads();

  // Phase 3: out = P @ V ; A from LDS, B from transposed V. Waves split d.
  const __bf16* Vtb = Vt + (size_t)b * DD * nK;  // [DD, nK]
  for (int dt = wave; dt < 16; dt += 4) {
    v8f acc = v8f{};
    for (int k0 = 0; k0 < nK; k0 += 32) {
      v16bf a  = frag_a_bf16((const __bf16*)S + k0, ldS);
      v16bf bb = frag_b_bf16(Vtb + (size_t)dt * 16 * nK + k0, nK);
      acc = wmma_bf16(a, bb, acc);
    }
    const int hi = lane >> 4;
    const int cn = lane & 15;
#pragma unroll
    for (int i = 0; i < 8; ++i) {
      const int r = i + 8 * hi;
      out[((size_t)b * nQ + q0 + r) * DD + dt * 16 + cn] = acc[i] * rowinv[r];
    }
  }
}

// out[b,d,r] = bf16(in[b,r,d])  — 32x32 LDS-tiled, both sides coalesced.
// grid: (rows/32, DD/32, B), block (32, 8).
__global__ __launch_bounds__(256) void transpose_bf16_tiled(
    const float* __restrict__ in, __bf16* __restrict__ out, int rows)
{
  __shared__ float tile[32][33];
  const int b  = blockIdx.z;
  const int r0 = blockIdx.x * 32;
  const int d0 = blockIdx.y * 32;
#pragma unroll
  for (int i = threadIdx.y; i < 32; i += 8)
    tile[i][threadIdx.x] =
        in[((size_t)b * rows + r0 + i) * DD + d0 + threadIdx.x];
  __syncthreads();
#pragma unroll
  for (int i = threadIdx.y; i < 32; i += 8)
    out[((size_t)b * DD + d0 + i) * rows + r0 + threadIdx.x] =
        (__bf16)tile[threadIdx.x][i];
}

__global__ void add_kernel(float* __restrict__ a, const float* __restrict__ b,
                           size_t n) {
  const size_t i = (size_t)blockIdx.x * blockDim.x + threadIdx.x;
  if (i < n) a[i] += b[i];
}

// ---------------------------------------------------------------------------
extern "C" void kernel_launch(void* const* d_in, const int* in_sizes, int n_in,
                              void* d_out, int out_size, void* d_ws, size_t ws_size,
                              hipStream_t stream)
{
  (void)in_sizes; (void)n_in; (void)out_size; (void)ws_size;
  const int B = 16, E = 1024, N = 2048, D = 256, L = 3;

  const float* ev_in   = (const float*)d_in[0];
  const float* ent_in  = (const float*)d_in[1];
  const float* ev2ent  = (const float*)d_in[2];  // [B,E,N]
  const float* ent2ev  = (const float*)d_in[3];  // [B,N,E]
  const float* w       = (const float*)d_in[4];  // [L,D,D]
  const float* wb      = (const float*)d_in[5];  // [L,D]
  const float* watt    = (const float*)d_in[6];  // [L,D,D]
  const float* wattb   = (const float*)d_in[7];  // [L,D]

  // outputs double as the running state (ent first, then ev)
  float* ent = (float*)d_out;
  float* ev  = ent + (size_t)B * N * D;

  // workspace: pe/pn/V^T bf16 + one fp32 msg buffer  (~76 MB)
  char* p = (char*)d_ws;
  __bf16* pe = (__bf16*)p; p += (size_t)B * E * D * sizeof(__bf16);
  __bf16* pn = (__bf16*)p; p += (size_t)B * N * D * sizeof(__bf16);
  __bf16* vT = (__bf16*)p; p += (size_t)B * D * N * sizeof(__bf16);
  float*  msg = (float*)p;

  hipMemcpyAsync(ent, ent_in, (size_t)B * N * D * sizeof(float),
                 hipMemcpyDeviceToDevice, stream);
  hipMemcpyAsync(ev, ev_in, (size_t)B * E * D * sizeof(float),
                 hipMemcpyDeviceToDevice, stream);

  const dim3 blk(128);
  const dim3 tblk(32, 8);
  for (int l = 0; l < L; ++l) {
    const float* Wl  = w     + (size_t)l * D * D;
    const float* bl  = wb    + (size_t)l * D;
    const float* Wal = watt  + (size_t)l * D * D;
    const float* bal = wattb + (size_t)l * D;

    // pe = bf16(ev @ Watt^T + b), pn = bf16(ent @ Watt^T + b)
    linear_wmma<true, false><<<dim3((B * E) / 64, D / 64), blk, 0, stream>>>(
        ev, Wal, bal, 1.f, nullptr, pe, B * E);
    linear_wmma<true, false><<<dim3((B * N) / 64, D / 64), blk, 0, stream>>>(
        ent, Wal, bal, 1.f, nullptr, pn, B * N);

    // dir1: entities gather from evidences (V = old ev)
    transpose_bf16_tiled<<<dim3(E / 32, D / 32, B), tblk, 0, stream>>>(ev, vT, E);
    attn_wmma<<<dim3(N / 16, B), blk, 0, stream>>>(pn, pe, vT, ev2ent, msg, N, E);

    // ent = relu((msg + ent) @ W^T + 2b)
    { size_t n = (size_t)B * N * D;
      add_kernel<<<(unsigned)((n + 255) / 256), 256, 0, stream>>>(msg, ent, n); }
    linear_wmma<false, true><<<dim3((B * N) / 64, D / 64), blk, 0, stream>>>(
        msg, Wl, bl, 2.f, ent, nullptr, B * N);

    // dir2: evidences gather from UPDATED entities (V = new ent)
    transpose_bf16_tiled<<<dim3(N / 32, D / 32, B), tblk, 0, stream>>>(ent, vT, N);
    attn_wmma<<<dim3(E / 16, B), blk, 0, stream>>>(pe, pn, vT, ent2ev, msg, E, N);

    // ev = relu((msg + ev) @ W^T + 2b)
    { size_t n = (size_t)B * E * D;
      add_kernel<<<(unsigned)((n + 255) / 256), 256, 0, stream>>>(msg, ev, n); }
    linear_wmma<false, true><<<dim3((B * E) / 64, D / 64), blk, 0, stream>>>(
        msg, Wl, bl, 2.f, ev, nullptr, B * E);
  }
}
